// AdaptiveGAT_39676907888757
// MI455X (gfx1250) — matrix operand order repaired and checked by reading, not compiled, not run on hardware
//
#include <hip/hip_runtime.h>

typedef __attribute__((ext_vector_type(16))) _Float16 v16h;
typedef __attribute__((ext_vector_type(8)))  float    v8f;
typedef __attribute__((ext_vector_type(4)))  float    f4;

#define N_NODES 50000
#define N_EDGES 800000
#define NBATCH  64
#define NODE_DIM 64
#define EDGE_DIM 16
#define HID 128
#define NCLS 6

static inline int cdiv(int a, int b) { return (a + b - 1) / b; }

// ---- monotone float<->uint map for atomicMax-based segment max ----
__device__ __forceinline__ unsigned mapf(float x) {
  unsigned u = __float_as_uint(x);
  return (u & 0x80000000u) ? ~u : (u | 0x80000000u);
}
__device__ __forceinline__ float unmapf(unsigned u) {
  return __uint_as_float((u & 0x80000000u) ? (u & 0x7FFFFFFFu) : ~u);
}

// ---- CDNA5 async global->LDS copy (16B per lane) + wait ----
// LDS operand: low 32 bits of the generic shared-aperture address == LDS byte
// offset (ISA 10.2).
__device__ __forceinline__ void async_copy_b128(const void* gsrc, void* ldst) {
  unsigned loff = (unsigned)(size_t)ldst;
  asm volatile("global_load_async_to_lds_b128 %0, %1, off"
               :: "v"(loff), "v"(gsrc) : "memory");
}
__device__ __forceinline__ void async_wait0() {
  asm volatile("s_wait_asynccnt 0x0" ::: "memory");
}

// ---- one-shot: W[K,NCOL] f32 -> Wt[n*Kp+k] f16 (zero-pad k>=K) ----
__global__ void k_prep_w16(const float* __restrict__ W, _Float16* __restrict__ Wt,
                           int K, int NCOL, int Kp) {
  int idx = blockIdx.x * blockDim.x + threadIdx.x;
  if (idx >= NCOL * Kp) return;
  int n = idx / Kp, k = idx - n * Kp;
  Wt[idx] = (k < K) ? (_Float16)W[k * NCOL + n] : (_Float16)0.f;
}

// =====================================================================
// Generic WMMA GEMM: Out[M,NCOL] = act(A[M,K] @ W[K,NCOL] + bias)
// Weights arrive pre-transposed f16; staged to LDS via async b128 copies.
// 8 waves/block, 16 rows/wave.  ACT: 0=none, 1=relu
// =====================================================================
template <int K, int NCOL, int ACT>
__global__ void k_gemm(const float* __restrict__ A, const _Float16* __restrict__ Wt16,
                       const float* __restrict__ bias, float* __restrict__ Out, int M) {
  __shared__ _Float16 Wt[NCOL * K];   // transposed weights: Wt[n*K + k]
  const int tid = threadIdx.x;
  for (int idx = tid * 8; idx < NCOL * K; idx += 256 * 8)
    async_copy_b128(Wt16 + idx, &Wt[idx]);
  async_wait0();
  __syncthreads();

  const int lane = tid & 31;
  const int wave = tid >> 5;
  const int rowBase = blockIdx.x * 128 + wave * 16;

  // ---- A fragments (16x32 f16 layout): lane half selects K sub-block ----
  const int mA = lane & 15;
  int row = rowBase + mA;
  int rc = row < M ? row : (M - 1);
  const float* arow = A + (size_t)rc * K;
  const int kb = (lane >> 4) << 3;   // 0 or 8
  v16h a[K / 32];
#pragma unroll
  for (int ks = 0; ks < K / 32; ++ks) {
    const int k0 = ks * 32 + kb;
#pragma unroll
    for (int j = 0; j < 8; ++j) {
      a[ks][j]     = (_Float16)arow[k0 + j];
      a[ks][8 + j] = (_Float16)arow[k0 + 16 + j];
    }
  }

  const int nlane = lane & 15;
  const int kbB = (lane >> 4) << 3;  // 0 or 8 (B: 32x16 f16 layout)
  const int mC = (lane >> 4) << 3;   // C row offset for this lane half

#pragma unroll
  for (int nt = 0; nt < NCOL / 16; ++nt) {
    const int n = nt * 16 + nlane;
    v8f c = {};
#pragma unroll
    for (int ks = 0; ks < K / 32; ++ks) {
      const _Float16* bp = &Wt[n * K + ks * 32 + kbB];
      v16h b;
#pragma unroll
      for (int j = 0; j < 16; ++j) b[j] = bp[(j & 7) + ((j >> 3) << 4)];
      c = __builtin_amdgcn_wmma_f32_16x16x32_f16(false, a[ks], false, b,
                                                 (short)0, c, false, false);
    }
    const float bv = bias ? bias[n] : 0.f;
#pragma unroll
    for (int r = 0; r < 8; ++r) {
      int orow = rowBase + mC + r;
      if (orow < M) {
        float v = c[r] + bv;
        if (ACT == 1) v = v > 0.f ? v : 0.f;
        Out[(size_t)orow * NCOL + n] = v;
      }
    }
  }
}

// =====================================================================
// Edge-weight MLP: w = alpha + (1-alpha)*sigmoid(relu(ea@W1+b1)@W2+b2)
// WMMA with K=16 zero-padded to 32; @W2 folded into half-wave shuffle reduce.
// Also accumulates sum(w) for the self-loop mean.
// =====================================================================
__global__ void k_edge_mlp(const float* __restrict__ EA, const _Float16* __restrict__ W1t,
                           const float* __restrict__ b1, const float* __restrict__ W2,
                           const float* __restrict__ b2, const float* __restrict__ alphaLogit,
                           float* __restrict__ wOut, float* __restrict__ wSum, int E) {
  __shared__ _Float16 Wt[64 * 32];   // Wt[n*32+k], zero-padded for k>=16
  const int tid = threadIdx.x;
  for (int idx = tid * 8; idx < 64 * 32; idx += 256 * 8)
    async_copy_b128(W1t + idx, &Wt[idx]);
  async_wait0();
  __syncthreads();

  const int lane = tid & 31;
  const int wave = tid >> 5;
  const int rowBase = (blockIdx.x * 8 + wave) * 16;
  const int mA = lane & 15;
  int row = rowBase + mA;
  int rc = row < E ? row : (E - 1);
  const float* arow = EA + (size_t)rc * EDGE_DIM;
  const int kb = (lane >> 4) << 3;
  v16h a;
#pragma unroll
  for (int j = 0; j < 8; ++j) {
    a[j]     = (_Float16)arow[kb + j];   // K = kb..kb+7  (covers 0..15)
    a[8 + j] = (_Float16)0.f;            // K >= 16 is padding
  }

  const int nlane = lane & 15;
  const int kbB = (lane >> 4) << 3;
  float p[8];
#pragma unroll
  for (int r = 0; r < 8; ++r) p[r] = 0.f;

#pragma unroll
  for (int nt = 0; nt < 4; ++nt) {
    const int n = nt * 16 + nlane;
    const _Float16* bp = &Wt[n * 32 + kbB];
    v16h b;
#pragma unroll
    for (int j = 0; j < 16; ++j) b[j] = bp[(j & 7) + ((j >> 3) << 4)];
    v8f c = {};
    c = __builtin_amdgcn_wmma_f32_16x16x32_f16(false, a, false, b,
                                               (short)0, c, false, false);
    const float bb = b1[n];
    const float w2 = W2[n];
#pragma unroll
    for (int r = 0; r < 8; ++r) {
      float hm = c[r] + bb;
      hm = hm > 0.f ? hm : 0.f;
      p[r] += hm * w2;
    }
  }
  // reduce columns within each 16-lane half (rows 0-7 low half, 8-15 high half)
#pragma unroll
  for (int off = 1; off < 16; off <<= 1) {
#pragma unroll
    for (int r = 0; r < 8; ++r) p[r] += __shfl_xor(p[r], off, 32);
  }
  const float alpha = 1.f / (1.f + __expf(-alphaLogit[0]));
  const float b2v = b2[0];
  if ((lane & 15) == 0) {
    const int mc = (lane >> 4) << 3;
    float acc = 0.f;
#pragma unroll
    for (int r = 0; r < 8; ++r) {
      int er = rowBase + mc + r;
      if (er < E) {
        float lw = 1.f / (1.f + __expf(-(p[r] + b2v)));
        float wv = alpha + (1.f - alpha) * lw;
        wOut[er] = wv;
        acc += wv;
      }
    }
    atomicAdd(wSum, acc);
  }
}

// ---- per-layer scalar: ae_coef[h] = sum_c lin_edge[h*C+c]*att_edge[h*C+c] ----
__global__ void k_att_coef(const float* __restrict__ lin_edge,
                           const float* __restrict__ att_edge,
                           float* __restrict__ aecoef, int H, int C) {
  int h = threadIdx.x;
  if (h < H) {
    float s = 0.f;
    for (int c = 0; c < C; ++c) s += lin_edge[h * C + c] * att_edge[h * C + c];
    aecoef[h] = s;
  }
}

// ---- a_src/a_dst per (node, head) ----
__global__ void k_node_att(const float* __restrict__ xh, const float* __restrict__ attS,
                           const float* __restrict__ attD, float* __restrict__ aS,
                           float* __restrict__ aD, int Nn, int H, int C) {
  int i = blockIdx.x * blockDim.x + threadIdx.x;
  if (i >= Nn * H) return;
  int n = i / H, h = i - n * H;
  const float* xp = xh + (size_t)n * HID + h * C;
  const float* ws = attS + h * C;
  const float* wd = attD + h * C;
  float s = 0.f, d = 0.f;
  for (int c = 0; c < C; ++c) {
    float v = xp[c];
    s += v * ws[c];
    d += v * wd[c];
  }
  aS[i] = s;
  aD[i] = d;
}

// ---- softmax pass 1: leaky-relu logits + segment max (mapped-uint atomicMax) ----
template <int H>
__global__ void k_edge_logits(const int* __restrict__ srcI, const int* __restrict__ dstI,
                              const float* __restrict__ wE, const float* __restrict__ wSum,
                              const float* __restrict__ aS, const float* __restrict__ aD,
                              const float* __restrict__ aecoef, float* __restrict__ att,
                              unsigned* __restrict__ segmax, int E, int Nn) {
  int e = blockIdx.x * blockDim.x + threadIdx.x;
  int ET = E + Nn;
  if (e >= ET) return;
  int s, d;
  float w;
  if (e < E) { s = srcI[e]; d = dstI[e]; w = wE[e]; }
  else       { s = d = e - E; w = wSum[0] * (1.f / (float)E); }
#pragma unroll
  for (int h = 0; h < H; ++h) {
    float l = aS[s * H + h] + aD[d * H + h] + w * aecoef[h];
    l = l > 0.f ? l : 0.2f * l;
    att[(size_t)e * H + h] = l;
    atomicMax(&segmax[d * H + h], mapf(l));
  }
}

// ---- softmax pass 2: exp + segment sum ----
template <int H>
__global__ void k_edge_exp(const int* __restrict__ dstI, float* __restrict__ att,
                           const unsigned* __restrict__ segmax, float* __restrict__ segsum,
                           int E, int Nn) {
  int e = blockIdx.x * blockDim.x + threadIdx.x;
  int ET = E + Nn;
  if (e >= ET) return;
  int d = (e < E) ? dstI[e] : (e - E);
#pragma unroll
  for (int h = 0; h < H; ++h) {
    float mx = unmapf(segmax[d * H + h]);
    float ex = __expf(att[(size_t)e * H + h] - mx);
    att[(size_t)e * H + h] = ex;
    atomicAdd(&segsum[d * H + h], ex);
  }
}

// ---- aggregation: wave per edge, 4 channels/lane, float atomics into agg ----
__global__ void k_aggregate(const int* __restrict__ srcI, const int* __restrict__ dstI,
                            const float* __restrict__ att, const float* __restrict__ segsum,
                            const float* __restrict__ xh, float* __restrict__ agg,
                            int E, int Nn, int H) {
  int gw = (blockIdx.x * blockDim.x + threadIdx.x) >> 5;
  int lane = threadIdx.x & 31;
  int ET = E + Nn;
  if (gw >= ET) return;
  int s, d;
  if (gw < E) { s = srcI[gw]; d = dstI[gw]; }
  else        { s = d = gw - E; }
  int c4 = lane * 4;
  int h = (c4 * H) >> 7;  // c4 / (128/H)
  float coef = att[(size_t)gw * H + h] / (segsum[d * H + h] + 1e-16f);
  const f4* xp = (const f4*)(xh + (size_t)s * HID);
  f4 v = xp[lane];
  float* op = agg + (size_t)d * HID + c4;
  atomicAdd(op + 0, coef * v[0]);
  atomicAdd(op + 1, coef * v[1]);
  atomicAdd(op + 2, coef * v[2]);
  atomicAdd(op + 3, coef * v[3]);
}

// ---- post: bias + layernorm + elu + residual, wave per node ----
__global__ void k_post(const float* __restrict__ agg, const float* __restrict__ bias,
                       const float* __restrict__ g, const float* __restrict__ b,
                       float* __restrict__ h, int Nn) {
  int gw = (blockIdx.x * blockDim.x + threadIdx.x) >> 5;
  int lane = threadIdx.x & 31;
  if (gw >= Nn) return;
  int c4 = lane * 4;
  float v[4];
#pragma unroll
  for (int i = 0; i < 4; ++i) v[i] = agg[(size_t)gw * HID + c4 + i] + bias[c4 + i];
  float sum = v[0] + v[1] + v[2] + v[3];
#pragma unroll
  for (int off = 1; off < 32; off <<= 1) sum += __shfl_xor(sum, off, 32);
  float mu = sum * (1.f / 128.f);
  float vs = 0.f;
#pragma unroll
  for (int i = 0; i < 4; ++i) { float dv = v[i] - mu; vs += dv * dv; }
#pragma unroll
  for (int off = 1; off < 32; off <<= 1) vs += __shfl_xor(vs, off, 32);
  float inv = rsqrtf(vs * (1.f / 128.f) + 1e-5f);
#pragma unroll
  for (int i = 0; i < 4; ++i) {
    float y = (v[i] - mu) * inv * g[c4 + i] + b[c4 + i];
    y = y > 0.f ? y : (__expf(y) - 1.f);  // elu
    h[(size_t)gw * HID + c4 + i] = y + h[(size_t)gw * HID + c4 + i];
  }
}

// ---- gate logit: dot(g1[n], W2) + b2, atomic batch max ----
__global__ void k_gate_logit(const float* __restrict__ g1, const float* __restrict__ W2,
                             const float* __restrict__ b2, const int* __restrict__ batch,
                             float* __restrict__ glog, unsigned* __restrict__ bmax, int Nn) {
  int n = blockIdx.x * blockDim.x + threadIdx.x;
  if (n >= Nn) return;
  float s = b2[0];
  const float* gp = g1 + (size_t)n * 64;
  for (int j = 0; j < 64; ++j) s += gp[j] * W2[j];
  glog[n] = s;
  atomicMax(&bmax[batch[n]], mapf(s));
}

__global__ void k_gate_exp(const float* __restrict__ glog, const int* __restrict__ batch,
                           const unsigned* __restrict__ bmax, float* __restrict__ gexp,
                           float* __restrict__ bsum, int Nn) {
  int n = blockIdx.x * blockDim.x + threadIdx.x;
  if (n >= Nn) return;
  int bI = batch[n];
  float ex = __expf(glog[n] - unmapf(bmax[bI]));
  gexp[n] = ex;
  atomicAdd(&bsum[bI], ex);
}

// ---- attention pooling: wave per node ----
__global__ void k_pool(const float* __restrict__ gexp, const float* __restrict__ bsum,
                       const int* __restrict__ batch, const float* __restrict__ h,
                       float* __restrict__ pooled, int Nn) {
  int gw = (blockIdx.x * blockDim.x + threadIdx.x) >> 5;
  int lane = threadIdx.x & 31;
  if (gw >= Nn) return;
  int bI = batch[gw];
  float coef = gexp[gw] / (bsum[bI] + 1e-16f);
  int c4 = lane * 4;
  const f4* hp = (const f4*)(h + (size_t)gw * HID);
  f4 v = hp[lane];
  float* op = pooled + (size_t)bI * HID + c4;
  atomicAdd(op + 0, coef * v[0]);
  atomicAdd(op + 1, coef * v[1]);
  atomicAdd(op + 2, coef * v[2]);
  atomicAdd(op + 3, coef * v[3]);
}

// ---- final classifier output: [B,64] @ [64,6] + b ----
__global__ void k_cls_out(const float* __restrict__ c1, const float* __restrict__ W2,
                          const float* __restrict__ b2, float* __restrict__ out) {
  int i = blockIdx.x * blockDim.x + threadIdx.x;
  if (i >= NBATCH * NCLS) return;
  int bI = i / NCLS, k = i - bI * NCLS;
  float s = b2[k];
  const float* cp = c1 + bI * 64;
  for (int j = 0; j < 64; ++j) s += cp[j] * W2[j * NCLS + k];
  out[i] = s;
}

// =====================================================================
extern "C" void kernel_launch(void* const* d_in, const int* in_sizes, int n_in,
                              void* d_out, int out_size, void* d_ws, size_t ws_size,
                              hipStream_t stream) {
  const float* x         = (const float*)d_in[0];
  const int*   edge_idx  = (const int*)d_in[1];
  const float* edge_attr = (const float*)d_in[2];
  const int*   batch_idx = (const int*)d_in[3];
  int pi = 4;
  auto F = [&](int i) { return (const float*)d_in[i]; };
  const float* node_W = F(pi++); const float* node_b = F(pi++);
  const float* ew_W1  = F(pi++); const float* ew_b1  = F(pi++);
  const float* ew_W2  = F(pi++); const float* ew_b2  = F(pi++);
  const float* alpha_logit = F(pi++);
  struct Layer { const float *lin, *lin_edge, *att_src, *att_dst, *att_edge, *bias, *ln_g, *ln_b; } L[3];
  for (int i = 0; i < 3; ++i) {
    L[i].lin = F(pi++); L[i].lin_edge = F(pi++);
    L[i].att_src = F(pi++); L[i].att_dst = F(pi++); L[i].att_edge = F(pi++);
    L[i].bias = F(pi++); L[i].ln_g = F(pi++); L[i].ln_b = F(pi++);
  }
  const float* gate_W1 = F(pi++); const float* gate_b1 = F(pi++);
  const float* gate_W2 = F(pi++); const float* gate_b2 = F(pi++);
  const float* cls_W1  = F(pi++); const float* cls_b1  = F(pi++);
  const float* cls_W2  = F(pi++); const float* cls_b2  = F(pi++);

  const int* srcI = edge_idx;            // edge_index[0]
  const int* dstI = edge_idx + N_EDGES;  // edge_index[1]
  const int ET = N_EDGES + N_NODES;

  // ---- workspace carve-out (256B aligned) ----
  char* wp = (char*)d_ws;
  auto alloc = [&](size_t bytes) { char* p = wp; wp += (bytes + 255) & ~(size_t)255; return p; };
  float*    h      = (float*)alloc((size_t)N_NODES * HID * 4);
  float*    xh     = (float*)alloc((size_t)N_NODES * HID * 4);
  float*    agg    = (float*)alloc((size_t)N_NODES * HID * 4);
  float*    aS     = (float*)alloc((size_t)N_NODES * 4 * 4);
  float*    aD     = (float*)alloc((size_t)N_NODES * 4 * 4);
  float*    wE     = (float*)alloc((size_t)N_EDGES * 4);
  float*    att    = (float*)alloc((size_t)ET * 4 * 4);
  unsigned* segmax = (unsigned*)alloc((size_t)N_NODES * 4 * 4);
  float*    segsum = (float*)alloc((size_t)N_NODES * 4 * 4);
  float*    aecoef = (float*)alloc(256);
  float*    wsum   = (float*)alloc(256);
  float*    glog   = (float*)alloc((size_t)N_NODES * 4);
  float*    gexp   = (float*)alloc((size_t)N_NODES * 4);
  unsigned* bmax   = (unsigned*)alloc(NBATCH * 4);
  float*    bsum   = (float*)alloc(NBATCH * 4);
  float*    pooled = (float*)alloc(NBATCH * HID * 4);
  float*    c1     = (float*)alloc(NBATCH * 64 * 4);
  // pre-transposed f16 weight images
  _Float16* nodeWt = (_Float16*)alloc((size_t)128 * 64 * 2);
  _Float16* linWt[3];
  for (int i = 0; i < 3; ++i) linWt[i] = (_Float16*)alloc((size_t)128 * 128 * 2);
  _Float16* gateWt = (_Float16*)alloc((size_t)64 * 128 * 2);
  _Float16* clsWt  = (_Float16*)alloc((size_t)64 * 128 * 2);
  _Float16* ewWt   = (_Float16*)alloc((size_t)64 * 32 * 2);
  float*    g1     = xh;  // reuse xh region after last GAT layer

  // ---- one-shot weight transpose/convert (f32 -> f16, n-major) ----
  k_prep_w16<<<cdiv(128 * 64, 256), 256, 0, stream>>>(node_W, nodeWt, 64, 128, 64);
  for (int i = 0; i < 3; ++i)
    k_prep_w16<<<cdiv(128 * 128, 256), 256, 0, stream>>>(L[i].lin, linWt[i], 128, 128, 128);
  k_prep_w16<<<cdiv(64 * 128, 256), 256, 0, stream>>>(gate_W1, gateWt, 128, 64, 128);
  k_prep_w16<<<cdiv(64 * 128, 256), 256, 0, stream>>>(cls_W1, clsWt, 128, 64, 128);
  k_prep_w16<<<cdiv(64 * 32, 256), 256, 0, stream>>>(ew_W1, ewWt, 16, 64, 32);

  // ---- node embedding + edge-weight MLP ----
  hipMemsetAsync(wsum, 0, 4, stream);
  k_gemm<NODE_DIM, HID, 1><<<cdiv(N_NODES, 128), 256, 0, stream>>>(x, nodeWt, node_b, h, N_NODES);
  k_edge_mlp<<<cdiv(N_EDGES, 128), 256, 0, stream>>>(edge_attr, ewWt, ew_b1, ew_W2, ew_b2,
                                                     alpha_logit, wE, wsum, N_EDGES);

  // ---- 3 GAT layers ----
  for (int i = 0; i < 3; ++i) {
    const int H = (i < 2) ? 4 : 1;
    const int C = HID / H;
    k_gemm<HID, HID, 0><<<cdiv(N_NODES, 128), 256, 0, stream>>>(h, linWt[i], nullptr, xh, N_NODES);
    k_att_coef<<<1, 32, 0, stream>>>(L[i].lin_edge, L[i].att_edge, aecoef, H, C);
    k_node_att<<<cdiv(N_NODES * H, 256), 256, 0, stream>>>(xh, L[i].att_src, L[i].att_dst,
                                                           aS, aD, N_NODES, H, C);
    hipMemsetAsync(segmax, 0, (size_t)N_NODES * H * 4, stream);
    hipMemsetAsync(segsum, 0, (size_t)N_NODES * H * 4, stream);
    hipMemsetAsync(agg, 0, (size_t)N_NODES * HID * 4, stream);
    if (H == 4) {
      k_edge_logits<4><<<cdiv(ET, 256), 256, 0, stream>>>(srcI, dstI, wE, wsum, aS, aD,
                                                          aecoef, att, segmax, N_EDGES, N_NODES);
      k_edge_exp<4><<<cdiv(ET, 256), 256, 0, stream>>>(dstI, att, segmax, segsum, N_EDGES, N_NODES);
    } else {
      k_edge_logits<1><<<cdiv(ET, 256), 256, 0, stream>>>(srcI, dstI, wE, wsum, aS, aD,
                                                          aecoef, att, segmax, N_EDGES, N_NODES);
      k_edge_exp<1><<<cdiv(ET, 256), 256, 0, stream>>>(dstI, att, segmax, segsum, N_EDGES, N_NODES);
    }
    k_aggregate<<<cdiv(ET, 8), 256, 0, stream>>>(srcI, dstI, att, segsum, xh, agg,
                                                 N_EDGES, N_NODES, H);
    k_post<<<cdiv(N_NODES, 8), 256, 0, stream>>>(agg, L[i].bias, L[i].ln_g, L[i].ln_b, h, N_NODES);
  }

  // ---- gated attention pooling over batches ----
  k_gemm<HID, 64, 1><<<cdiv(N_NODES, 128), 256, 0, stream>>>(h, gateWt, gate_b1, g1, N_NODES);
  hipMemsetAsync(bmax, 0, NBATCH * 4, stream);
  hipMemsetAsync(bsum, 0, NBATCH * 4, stream);
  hipMemsetAsync(pooled, 0, NBATCH * HID * 4, stream);
  k_gate_logit<<<cdiv(N_NODES, 256), 256, 0, stream>>>(g1, gate_W2, gate_b2, batch_idx,
                                                       glog, bmax, N_NODES);
  k_gate_exp<<<cdiv(N_NODES, 256), 256, 0, stream>>>(glog, batch_idx, bmax, gexp, bsum, N_NODES);
  k_pool<<<cdiv(N_NODES, 8), 256, 0, stream>>>(gexp, bsum, batch_idx, h, pooled, N_NODES);

  // ---- classifier ----
  k_gemm<HID, 64, 1><<<1, 256, 0, stream>>>(pooled, clsWt, cls_b1, c1, NBATCH);
  k_cls_out<<<cdiv(NBATCH * NCLS, 256), 256, 0, stream>>>(c1, cls_W2, cls_b2, (float*)d_out);
}
